// Attention_17145509445886
// MI455X (gfx1250) — matrix operand order repaired and checked by reading, not compiled, not run on hardware
//
#include <hip/hip_runtime.h>
#include <hip/hip_bf16.h>
#include <math.h>

// ---------------------------------------------------------------------------
// Fused attention for (B,S,N,D)=(4,1024,16,512), H=8, HD=64 on gfx1250.
// One block per (b,s): 16x512 panel. bf16 WMMA (16x16x32) with f32 accum.
// Prep kernel converts W_qkv/W_out to bf16 *transposed* ([n][k]) so the WMMA
// B operand is two 16B per-lane loads from one [n] row, and builds rotary cos/sin LUTs.
// ---------------------------------------------------------------------------

typedef __attribute__((ext_vector_type(16))) __bf16 v16bf;
typedef __attribute__((ext_vector_type(8)))  float  v8f;

#define WAVES 8
#define BLOCK 256

static constexpr int Bb = 4, Ss = 1024, Nn = 16, Dd = 512, Hh = 8, HDd = 64;
static constexpr int QKVN = 3 * Hh * HDd;   // 1536

union V16 { v16bf v; int4 q[2]; };

__device__ __forceinline__ v16bf ld2x16B(const __bf16* p0, const __bf16* p1) {
    V16 u;
    u.q[0] = *(const int4*)p0;   // 16B -> ds_load_b128 / global_load_b128
    u.q[1] = *(const int4*)p1;
    return u.v;
}

// ------------------------- prep: weights -> bf16^T, rotary LUTs -------------
__global__ void prep_kernel(const float* __restrict__ Wqkv,
                            const float* __restrict__ Wout,
                            __bf16* __restrict__ WqkvT,   // [1536][512]
                            __bf16* __restrict__ WoutT,   // [512][512]
                            float* __restrict__ costab,   // [16][64]
                            float* __restrict__ sintab) {
    const int nQ = QKVN * Dd;          // 786432
    const int nO = Dd * Dd;            // 262144
    int idx = blockIdx.x * blockDim.x + threadIdx.x;
    if (idx < nQ) {
        int n = idx / Dd, k = idx - n * Dd;
        WqkvT[idx] = (__bf16)Wqkv[k * QKVN + n];
    } else if (idx < nQ + nO) {
        int j = idx - nQ;
        int n = j / Dd, k = j - n * Dd;
        WoutT[j] = (__bf16)Wout[k * Dd + n];
    } else if (idx < nQ + nO + Nn * HDd) {
        int j = idx - nQ - nO;
        int n = j / HDd, d = j - n * HDd;
        float inv = __powf(10000.0f, -(float)(2 * (d >> 1)) / (float)HDd);
        float f = (float)n * inv;
        costab[j] = __cosf(f);
        sintab[j] = __sinf(f);
    }
}

// ------------------------------ fused kernel --------------------------------
__global__ __launch_bounds__(BLOCK)
void attn_fused_kernel(const float* __restrict__ x,
                       const float* __restrict__ pos_bias,       // [8][16][16]
                       const unsigned char* __restrict__ focus,  // [4] bool
                       const __bf16* __restrict__ WqkvT,         // [1536][512]
                       const __bf16* __restrict__ WoutT,         // [512][512]
                       const float* __restrict__ costab,
                       const float* __restrict__ sintab,
                       float* __restrict__ out) {
    __shared__ __align__(16) __bf16 bufX [16 * 512];   // x panel, later head-outs
    __shared__ __align__(16) __bf16 bufQK[16 * 1024];  // q: cols 0..511, k: 512..1023
    __shared__ __align__(16) __bf16 bufVT[512 * 16];   // v transposed: [hd][j]
    __shared__ __align__(16) __bf16 attnS[8 * 256];    // per-head 16x16 probs

    const int blk  = blockIdx.x;          // b*S + s
    const int b    = blk >> 10;
    const int tid  = threadIdx.x;
    const int lane = tid & 31;
    const int wave = tid >> 5;
    const int m    = lane & 15;           // A-row / B-col / D-col index
    const int hi   = lane >> 4;           // half-wave select

    // ---- phase 0: load x panel (16x512 f32, coalesced) -> bf16 LDS ----
    const float4* xsrc = (const float4*)(x + (size_t)blk * (16 * 512));
    for (int i = tid; i < 16 * 512 / 4; i += BLOCK) {
        float4 f = xsrc[i];
        __bf16* d = &bufX[i * 4];
        d[0] = (__bf16)f.x; d[1] = (__bf16)f.y; d[2] = (__bf16)f.z; d[3] = (__bf16)f.w;
    }
    __syncthreads();

    // ---- phase 1: qkv = X @ Wqkv (M16 K512 N1536), 96 tiles over 8 waves ----
    for (int t = wave; t < 96; t += WAVES) {
        const int col0 = t * 16;
        v8f acc = {0.f,0.f,0.f,0.f,0.f,0.f,0.f,0.f};
        const __bf16* brow = WqkvT + (size_t)(col0 + m) * 512 + hi * 8;
        #pragma unroll 4
        for (int k0 = 0; k0 < 512; k0 += 32) {
            v16bf a  = ld2x16B(&bufX[m * 512 + k0 + hi * 8],
                               &bufX[m * 512 + k0 + 16 + hi * 8]);
            v16bf bm = ld2x16B(brow + k0, brow + k0 + 16);
            acc = __builtin_amdgcn_wmma_f32_16x16x32_bf16(false, a, false, bm,
                                                          (short)0, acc, false, false);
        }
        if (col0 < 1024) {                       // q or k: scale (q) + rotary
            const bool is_q = col0 < 512;
            const int  hd   = (col0 & 63) + m;   // head-dim index (col of D tile)
            const float sgn = (hd & 1) ? 1.0f : -1.0f;
            #pragma unroll
            for (int r = 0; r < 8; ++r) {
                int row = r + 8 * hi;            // position n
                float vv = acc[r];
                if (is_q) vv *= 0.125f;          // HD^-0.5
                float part = __shfl_xor(vv, 1, 32);   // partner feature (hd^1)
                float rot  = vv * costab[row * 64 + hd]
                           + sgn * part * sintab[row * 64 + hd];
                bufQK[row * 1024 + col0 + m] = (__bf16)rot;
            }
        } else {                                 // v: store transposed [hd][j]
            const int vc0 = col0 - 1024;
            #pragma unroll
            for (int r = 0; r < 8; ++r)
                bufVT[(vc0 + m) * 16 + (r + 8 * hi)] = (__bf16)acc[r];
        }
    }
    __syncthreads();

    // ---- phase 2: per-head attention, wave w <-> head h=w ----
    {
        const int h = wave;
        const bool fp = focus[b] != 0;
        v8f s = {0.f,0.f,0.f,0.f,0.f,0.f,0.f,0.f};
        #pragma unroll
        for (int k0 = 0; k0 < 64; k0 += 32) {    // sim = q_h @ k_h^T
            v16bf a  = ld2x16B(&bufQK[m * 1024 + h * 64 + k0 + hi * 8],
                               &bufQK[m * 1024 + h * 64 + k0 + 16 + hi * 8]);
            v16bf bm = ld2x16B(&bufQK[m * 1024 + 512 + h * 64 + k0 + hi * 8],
                               &bufQK[m * 1024 + 512 + h * 64 + k0 + 16 + hi * 8]);
            s = __builtin_amdgcn_wmma_f32_16x16x32_bf16(false, a, false, bm,
                                                        (short)0, s, false, false);
        }
        const float* pb = pos_bias + h * 256;
        #pragma unroll
        for (int r = 0; r < 8; ++r) {            // bias + mask + 16-wide softmax
            int i = r + 8 * hi;
            float val = s[r] + pb[i * 16 + m];
            if (fp && (i != m)) val = -3.402823466e38f;
            float mx = val;
            #pragma unroll
            for (int d = 8; d >= 1; d >>= 1) mx = fmaxf(mx, __shfl_xor(mx, d, 32));
            float e = __expf(val - mx);
            float sm = e;
            #pragma unroll
            for (int d = 8; d >= 1; d >>= 1) sm += __shfl_xor(sm, d, 32);
            attnS[h * 256 + i * 16 + m] = (__bf16)(e / sm);
        }
    }
    __syncthreads();

    // ---- phase 2b: out_h = attn @ v_h (K=16 zero-padded to 32) ----
    {
        const int h = wave;
        V16 ua;
        ua.q[0] = *(const int4*)&attnS[h * 256 + m * 16 + hi * 8];  // k = hi*8..+7 (<16)
        ua.q[1] = int4{0, 0, 0, 0};                                  // k >= 16 -> 0
        #pragma unroll
        for (int t = 0; t < 4; ++t) {
            const int c0 = t * 16;
            V16 ub;
            {                // k = hi*8..+7 valid rows of v; k >= 16 -> zero
                const __bf16* vp = &bufVT[(h * 64 + c0 + m) * 16 + hi * 8];
                ub.q[0] = *(const int4*)vp;
                ub.q[1] = int4{0,0,0,0};
            }
            v8f o = {0.f,0.f,0.f,0.f,0.f,0.f,0.f,0.f};
            o = __builtin_amdgcn_wmma_f32_16x16x32_bf16(false, ua.v, false, ub.v,
                                                        (short)0, o, false, false);
            #pragma unroll
            for (int r = 0; r < 8; ++r)
                bufX[(r + 8 * hi) * 512 + h * 64 + c0 + m] = (__bf16)o[r];
        }
    }
    __syncthreads();

    // ---- phase 3: y = heads @ W_out (M16 K512 N512), 32 tiles / 8 waves ----
    float* orow = out + (size_t)blk * (16 * 512);
    for (int t = wave; t < 32; t += WAVES) {
        const int col0 = t * 16;
        v8f acc = {0.f,0.f,0.f,0.f,0.f,0.f,0.f,0.f};
        const __bf16* brow = WoutT + (size_t)(col0 + m) * 512 + hi * 8;
        #pragma unroll 4
        for (int k0 = 0; k0 < 512; k0 += 32) {
            v16bf a  = ld2x16B(&bufX[m * 512 + k0 + hi * 8],
                               &bufX[m * 512 + k0 + 16 + hi * 8]);
            v16bf bm = ld2x16B(brow + k0, brow + k0 + 16);
            acc = __builtin_amdgcn_wmma_f32_16x16x32_bf16(false, a, false, bm,
                                                          (short)0, acc, false, false);
        }
        #pragma unroll
        for (int r = 0; r < 8; ++r)
            orow[(r + 8 * hi) * 512 + col0 + m] = acc[r];
    }
}

// ---------------------------------------------------------------------------
extern "C" void kernel_launch(void* const* d_in, const int* in_sizes, int n_in,
                              void* d_out, int out_size, void* d_ws, size_t ws_size,
                              hipStream_t stream) {
    const float* x        = (const float*)d_in[0];
    const float* pos_bias = (const float*)d_in[1];
    const float* Wqkv     = (const float*)d_in[2];
    const float* Wout     = (const float*)d_in[3];
    const unsigned char* focus = (const unsigned char*)d_in[4];

    char* ws = (char*)d_ws;                         // ~2.06 MB used
    __bf16* WqkvT = (__bf16*)ws;                                    // 1536*512*2
    __bf16* WoutT = (__bf16*)(ws + (size_t)QKVN * Dd * 2);          // 512*512*2
    float*  costab = (float*)(ws + (size_t)QKVN * Dd * 2 + (size_t)Dd * Dd * 2);
    float*  sintab = costab + Nn * HDd;

    const int prepN = QKVN * Dd + Dd * Dd + Nn * HDd;
    prep_kernel<<<(prepN + 255) / 256, 256, 0, stream>>>(Wqkv, Wout, WqkvT, WoutT,
                                                         costab, sintab);
    attn_fused_kernel<<<Bb * Ss, BLOCK, 0, stream>>>(x, pos_bias, focus, WqkvT, WoutT,
                                                     costab, sintab, (float*)d_out);
}
